// Temporal_GAT_Transformer_73667279061280
// MI455X (gfx1250) — matrix operand order repaired and checked by reading, not compiled, not run on hardware
//
#include <hip/hip_runtime.h>
#include <math.h>

// ---------------------------------------------------------------------------
// Temporal GAT + Transformer for MI455X (gfx1250, wave32, WMMA).
// All GEMMs go through v_wmma_f32_16x16x32_bf16 (bf16 in, f32 accumulate).
// Fragment loads are unguarded float4 (b128) vector loads (all M/N multiples
// of 16; K multiple of 32 or compile-time-16). f32->bf16 uses the native cast
// (RNE fptrunc -> hw cvt); wave indices go through readfirstlane so
// wave-uniform loops use scalar branches instead of exec masking.
//
// Param flattening assumption: d_in = [x, edge_index, <params leaves in
// recursive dict-insertion order>] (94 inputs total).
// ---------------------------------------------------------------------------

#define B_   2
#define N_   256
#define T_   8
#define IND_ 32
#define D_   64
#define E_   8192
#define NEG_SLOPE_ 0.2f

typedef __bf16 bf16_t;
typedef bf16_t v16bf __attribute__((ext_vector_type(16)));
typedef float  v8f   __attribute__((ext_vector_type(8)));

__device__ __forceinline__ bf16_t f2bf(float f) {
#if defined(__gfx1250__) || defined(__AMDGCN__)
  return (bf16_t)f;                       // native cvt, round-to-nearest-even
#else
  unsigned u = __float_as_uint(f);
  unsigned r = u + 0x7FFFu + ((u >> 16) & 1u);
  unsigned short hs = (unsigned short)(r >> 16);
  bf16_t o;
  __builtin_memcpy(&o, &hs, 2);
  return o;
#endif
}

__device__ __forceinline__ v8f zero8() {
  v8f z;
#pragma unroll
  for (int i = 0; i < 8; ++i) z[i] = 0.f;
  return z;
}

__device__ __forceinline__ v8f wmma_bf16(v16bf a, v16bf b, v8f c) {
  return __builtin_amdgcn_wmma_f32_16x16x32_bf16(false, a, false, b,
                                                 (short)0, c, false, false);
}

// A fragment: 16x32 tile of row-major A (lda), rows m0.., K cols k0..k0+31.
// Layout: row M=lane&15; lanes 0-15 elems = K 0..15, lanes 16-31 = K 16..31.
// K16: lanes 16-31 (K 16..31) are compile-time zero.
template <bool K16>
__device__ __forceinline__ v16bf load_a_frag(const float* __restrict__ A, int lda,
                                             int m0, int k0, int lane) {
  int row = m0 + (lane & 15);
  int koff = (lane & 16) ? 16 : 0;
  v16bf a;
  if (K16 && (lane & 16)) {
#pragma unroll
    for (int e = 0; e < 16; ++e) a[e] = f2bf(0.f);
    return a;
  }
  const float* Ar = A + (size_t)row * lda + k0 + koff;
  float4 p0 = *(const float4*)(Ar);
  float4 p1 = *(const float4*)(Ar + 4);
  float4 p2 = *(const float4*)(Ar + 8);
  float4 p3 = *(const float4*)(Ar + 12);
  a[0] = f2bf(p0.x); a[1] = f2bf(p0.y); a[2] = f2bf(p0.z); a[3] = f2bf(p0.w);
  a[4] = f2bf(p1.x); a[5] = f2bf(p1.y); a[6] = f2bf(p1.z); a[7] = f2bf(p1.w);
  a[8]  = f2bf(p2.x); a[9]  = f2bf(p2.y); a[10] = f2bf(p2.z); a[11] = f2bf(p2.w);
  a[12] = f2bf(p3.x); a[13] = f2bf(p3.y); a[14] = f2bf(p3.z); a[15] = f2bf(p3.w);
  return a;
}

// B fragment where B = W^T, W row-major (N x K, ldw): B[k, col] = W[col, k].
// Contiguous-K layout (same as load_a_frag): col = lane&15; lanes 0-15 elems = K 0..15, lanes 16-31 = K 16..31.
// K16: lanes 16-31 (K 16..31) are compile-time zero.
template <bool K16>
__device__ __forceinline__ v16bf load_bT_frag(const float* __restrict__ W, int ldw,
                                              int n0, int k0, int lane) {
  int col = n0 + (lane & 15);
  v16bf b;
  if (K16 && (lane & 16)) {
#pragma unroll
    for (int e = 0; e < 16; ++e) b[e] = f2bf(0.f);
    return b;
  }
  const float* Wr = W + (size_t)col * ldw + k0 + ((lane & 16) ? 16 : 0);
  float4 p0 = *(const float4*)(Wr);
  float4 p1 = *(const float4*)(Wr + 4);
  float4 p2 = *(const float4*)(Wr + 8);
  float4 p3 = *(const float4*)(Wr + 12);
  b[0]  = f2bf(p0.x); b[1]  = f2bf(p0.y); b[2]  = f2bf(p0.z); b[3]  = f2bf(p0.w);
  b[4]  = f2bf(p1.x); b[5]  = f2bf(p1.y); b[6]  = f2bf(p1.z); b[7]  = f2bf(p1.w);
  b[8]  = f2bf(p2.x); b[9]  = f2bf(p2.y); b[10] = f2bf(p2.z); b[11] = f2bf(p2.w);
  b[12] = f2bf(p3.x); b[13] = f2bf(p3.y); b[14] = f2bf(p3.z); b[15] = f2bf(p3.w);
  return b;
}

// B fragment from row-major B (K x N, ldb): B[k, col] = Bm[k*ldb + col].
__device__ __forceinline__ v16bf load_b_rm_frag(const float* __restrict__ Bm, int ldb,
                                                int n0, int k0, int lane) {
  int col = n0 + (lane & 15);
  int kb = k0 + ((lane & 16) ? 16 : 0);
  const float* Bc = Bm + (size_t)kb * ldb + col;
  v16bf b;
#pragma unroll
  for (int e = 0; e < 16; ++e) b[e] = f2bf(Bc[(size_t)e * ldb]);
  return b;
}

// monotone float <-> unsigned key for atomic max on floats (incl. negatives)
__device__ __forceinline__ unsigned fkey(float f) {
  unsigned u = __float_as_uint(f);
  return (u & 0x80000000u) ? ~u : (u | 0x80000000u);
}
__device__ __forceinline__ float unfkey(unsigned k) {
  unsigned u = (k & 0x80000000u) ? (k ^ 0x80000000u) : ~k;
  return __uint_as_float(u);
}

// ---------------------------------------------------------------------------
// GEMM: C[M, NT*16] = act(A[M,KC] @ W[NT*16, KC]^T + bias).
// Wave owns a 16 x (NT*16) strip: A fragment loaded once per k-step and
// reused across NT B fragments (NT accumulators).
// M multiple of 16; KC in {16, 64}.
// ---------------------------------------------------------------------------
template <int KC, int NT>
__global__ __launch_bounds__(256)
void wmma_gemm_nt(const float* __restrict__ A, const float* __restrict__ W,
                  const float* __restrict__ bias, float* __restrict__ C,
                  int M, int relu) {
  constexpr int N = NT * 16;
  constexpr bool K16 = (KC == 16);
  int lane = threadIdx.x & 31;
  int wib = __builtin_amdgcn_readfirstlane(threadIdx.x >> 5);   // wave in block
  int wave = blockIdx.x * (blockDim.x >> 5) + wib;
  int nwaves = gridDim.x * (blockDim.x >> 5);
  int strips = M >> 4;
  for (int s = wave; s < strips; s += nwaves) {
    int m0 = s << 4;
    v8f acc[NT];
#pragma unroll
    for (int j = 0; j < NT; ++j) acc[j] = zero8();
#pragma unroll
    for (int k0 = 0; k0 < KC; k0 += 32) {
      v16bf a = load_a_frag<K16>(A, KC, m0, k0, lane);
#pragma unroll
      for (int j = 0; j < NT; ++j) {
        v16bf b = load_bT_frag<K16>(W, KC, j * 16, k0, lane);
        acc[j] = wmma_bf16(a, b, acc[j]);
      }
    }
    int cl = lane & 15;
    int rbase = m0 + ((lane & 16) ? 8 : 0);
#pragma unroll
    for (int j = 0; j < NT; ++j) {
      int col = j * 16 + cl;
      float bv = bias[col];
#pragma unroll
      for (int r = 0; r < 8; ++r) {
        float v = acc[j][r] + bv;
        if (relu) v = fmaxf(v, 0.f);
        C[(size_t)(rbase + r) * N + col] = v;
      }
    }
  }
}

// ---------------------------------------------------------------------------
// Attention: per (q-tile of 16, head, batch). Two-pass softmax, score strip
// (16 x Lk) staged in LDS, QK^T and P@V on WMMA. dh = 16, scale = 0.25.
// Lk is a multiple of 32.
// ---------------------------------------------------------------------------
__global__ __launch_bounds__(256)
void attn_kernel(const float* __restrict__ Q, const float* __restrict__ Km,
                 const float* __restrict__ V, float* __restrict__ O,
                 int Lq, int Lk, float scale) {
  extern __shared__ float smem_f[];
  float* S = smem_f;                  // 16 * Lk
  float* Opart = S + (size_t)16 * Lk; // 8 waves * 256

  int qb = blockIdx.x, hd = blockIdx.y, b = blockIdx.z;
  const float* Qb = Q + (size_t)b * Lq * 64 + hd * 16;
  const float* Kb = Km + (size_t)b * Lk * 64 + hd * 16;
  const float* Vb = V + (size_t)b * Lk * 64 + hd * 16;
  float* Ob = O + (size_t)b * Lq * 64 + hd * 16;

  int wave = __builtin_amdgcn_readfirstlane(threadIdx.x >> 5);
  int lane = threadIdx.x & 31;
  int col = lane & 15;
  int hi8 = (lane & 16) ? 8 : 0;

  // S = scale * Q_tile @ K^T   (K-dim 16, compile-time zero padded to 32)
  v16bf aq = load_a_frag<true>(Qb, 64, qb * 16, 0, lane);
  for (int kb = wave; kb < (Lk >> 4); kb += 8) {
    v16bf bk = load_bT_frag<true>(Kb, 64, kb * 16, 0, lane);
    v8f acc = wmma_bf16(aq, bk, zero8());
#pragma unroll
    for (int r = 0; r < 8; ++r)
      S[(size_t)(r + hi8) * Lk + kb * 16 + col] = acc[r] * scale;
  }
  __syncthreads();

  // row-wise softmax: 16 threads per row (groups stay inside a wave half)
  {
    int g = threadIdx.x >> 4;        // row 0..15
    int tl = threadIdx.x & 15;
    float* Sr = S + (size_t)g * Lk;
    float m = -INFINITY;
    for (int j = tl; j < Lk; j += 16) m = fmaxf(m, Sr[j]);
#pragma unroll
    for (int o = 8; o > 0; o >>= 1) m = fmaxf(m, __shfl_xor(m, o, 32));
    float l = 0.f;
    for (int j = tl; j < Lk; j += 16) {
      float p = __expf(Sr[j] - m);
      Sr[j] = p;
      l += p;
    }
#pragma unroll
    for (int o = 8; o > 0; o >>= 1) l += __shfl_xor(l, o, 32);
    float inv = 1.f / l;
    for (int j = tl; j < Lk; j += 16) Sr[j] *= inv;
  }
  __syncthreads();

  // O_tile = P @ V, K = Lk split across 8 waves in chunks of 32
  v8f acc = zero8();
  for (int c = wave; c < (Lk >> 5); c += 8) {
    v16bf a = load_a_frag<false>(S, Lk, 0, c * 32, lane);
    v16bf bv = load_b_rm_frag(Vb, 64, 0, c * 32, lane);
    acc = wmma_bf16(a, bv, acc);
  }
#pragma unroll
  for (int r = 0; r < 8; ++r)
    Opart[wave * 256 + (r + hi8) * 16 + col] = acc[r];
  __syncthreads();

  int row = threadIdx.x >> 4;
  int oc = threadIdx.x & 15;
  float s = 0.f;
#pragma unroll
  for (int w = 0; w < 8; ++w) s += Opart[w * 256 + row * 16 + oc];
  Ob[(size_t)(qb * 16 + row) * 64 + oc] = s;
}

// ---------------------------------------------------------------------------
// y = LayerNorm(x (+ r)) * g + b ; row width 64, one wave per row
// ---------------------------------------------------------------------------
__global__ __launch_bounds__(256)
void add_ln_kernel(const float* __restrict__ x, const float* __restrict__ r,
                   const float* __restrict__ g, const float* __restrict__ bta,
                   float* __restrict__ y, int rows) {
  int wib = __builtin_amdgcn_readfirstlane(threadIdx.x >> 5);
  int wave = blockIdx.x * (blockDim.x >> 5) + wib;
  int lane = threadIdx.x & 31;
  int nw = gridDim.x * (blockDim.x >> 5);
  for (int row = wave; row < rows; row += nw) {
    size_t base = (size_t)row * 64;
    float v0 = x[base + lane], v1 = x[base + lane + 32];
    if (r) { v0 += r[base + lane]; v1 += r[base + lane + 32]; }
    float s = v0 + v1, sq = v0 * v0 + v1 * v1;
#pragma unroll
    for (int o = 16; o > 0; o >>= 1) {
      s += __shfl_xor(s, o, 32);
      sq += __shfl_xor(sq, o, 32);
    }
    float mean = s * (1.f / 64.f);
    float var = sq * (1.f / 64.f) - mean * mean;
    float inv = rsqrtf(var + 1e-5f);
    y[base + lane]      = (v0 - mean) * inv * g[lane] + bta[lane];
    y[base + lane + 32] = (v1 - mean) * inv * g[lane + 32] + bta[lane + 32];
  }
}

// ---------------------------------------------------------------------------
// GAT: one block per (b,t) graph. z = x_g @ W^T via WMMA into LDS (wave owns
// a 16x128 strip, A-frag reused across 8 B tiles), then per-head edge softmax
// (segment max/sum via LDS atomics), scatter-add, head-mean + bias + pos
// encoding -> h (B,N,T,D); also extracts tgt (t = T-1).
// ---------------------------------------------------------------------------
__global__ __launch_bounds__(256)
void gat_kernel(const float* __restrict__ x, const int* __restrict__ ei,
                const float* __restrict__ W, const float* __restrict__ asrc,
                const float* __restrict__ adst, const float* __restrict__ bias,
                float* __restrict__ h, float* __restrict__ tgt) {
  extern __shared__ float smem_f[];
  float* z    = smem_f;              // 256*128
  float* outp = z + 256 * 128;       // 256*128
  float* es   = outp + 256 * 128;    // 512
  float* ed   = es + 512;            // 512
  float* ssum = ed + 512;            // 512
  unsigned* mk = (unsigned*)(ssum + 512); // 512

  int gi = blockIdx.x;
  int b = gi / T_, t = gi - b * T_;
  int tid = threadIdx.x;
  int wave = __builtin_amdgcn_readfirstlane(tid >> 5);
  int lane = tid & 31;

  // z = x_g (256x32) @ W^T (32x128): wave owns 16x128 strip, K=32 one step
  const float* Ax = x + ((size_t)b * N_) * T_ * IND_ + (size_t)t * IND_; // lda=T_*IND_
  for (int s = wave; s < 16; s += 8) {
    int m0 = s << 4;
    v16bf a = load_a_frag<false>(Ax, T_ * IND_, m0, 0, lane);
    int cl = lane & 15;
    int rbase = m0 + ((lane & 16) ? 8 : 0);
#pragma unroll
    for (int j = 0; j < 8; ++j) {
      v16bf bw = load_bT_frag<false>(W, IND_, j * 16, 0, lane);
      v8f acc = wmma_bf16(a, bw, zero8());
#pragma unroll
      for (int r = 0; r < 8; ++r)
        z[(size_t)(rbase + r) * 128 + j * 16 + cl] = acc[r];
    }
  }
  // init accumulators
  for (int i = tid; i < 256 * 128; i += 256) outp[i] = 0.f;
  for (int i = tid; i < 512; i += 256) {
    mk[i] = fkey(-INFINITY);
    ssum[i] = 0.f;
  }
  __syncthreads();

  // attention logits per (node, head)
  for (int item = tid; item < 512; item += 256) {
    int n = item >> 1, hh = item & 1;
    const float* zr = z + (size_t)n * 128 + hh * 64;
    float s0 = 0.f, s1 = 0.f;
    for (int f = 0; f < 64; ++f) {
      s0 += zr[f] * asrc[hh * 64 + f];
      s1 += zr[f] * adst[hh * 64 + f];
    }
    es[item] = s0;
    ed[item] = s1;
  }
  __syncthreads();

  const int ET = E_ + N_;
  // pass 1: segment max
  for (int e = tid; e < ET; e += 256) {
    int sN = (e < E_) ? ei[e] : (e - E_);
    int dN = (e < E_) ? ei[E_ + e] : (e - E_);
#pragma unroll
    for (int hh = 0; hh < 2; ++hh) {
      float v = es[sN * 2 + hh] + ed[dN * 2 + hh];
      v = (v > 0.f) ? v : NEG_SLOPE_ * v;
      atomicMax(&mk[dN * 2 + hh], fkey(v));
    }
  }
  __syncthreads();
  // pass 2: segment sum of exp
  for (int e = tid; e < ET; e += 256) {
    int sN = (e < E_) ? ei[e] : (e - E_);
    int dN = (e < E_) ? ei[E_ + e] : (e - E_);
#pragma unroll
    for (int hh = 0; hh < 2; ++hh) {
      float v = es[sN * 2 + hh] + ed[dN * 2 + hh];
      v = (v > 0.f) ? v : NEG_SLOPE_ * v;
      float ex = __expf(v - unfkey(mk[dN * 2 + hh]));
      atomicAdd(&ssum[dN * 2 + hh], ex);
    }
  }
  __syncthreads();
  // pass 3: alpha-weighted scatter-add of z[src]
  for (int e = tid; e < ET; e += 256) {
    int sN = (e < E_) ? ei[e] : (e - E_);
    int dN = (e < E_) ? ei[E_ + e] : (e - E_);
#pragma unroll
    for (int hh = 0; hh < 2; ++hh) {
      float v = es[sN * 2 + hh] + ed[dN * 2 + hh];
      v = (v > 0.f) ? v : NEG_SLOPE_ * v;
      float alpha = __expf(v - unfkey(mk[dN * 2 + hh])) / ssum[dN * 2 + hh];
      const float* zr = z + (size_t)sN * 128 + hh * 64;
      float* orow = outp + (size_t)dN * 128 + hh * 64;
      for (int f = 0; f < 64; ++f) atomicAdd(&orow[f], alpha * zr[f]);
    }
  }
  __syncthreads();

  // head mean + bias + positional encoding -> h[b,n,t,:] ; tgt at t = T-1
  const float kdiv = 0.1439116f; // ln(10000)/64
  for (int it = tid; it < N_ * D_; it += 256) {
    int n = it >> 6, f = it & 63;
    float val = 0.5f * (outp[(size_t)n * 128 + f] + outp[(size_t)n * 128 + 64 + f]) + bias[f];
    float div = __expf(-(float)((f >> 1) << 1) * kdiv);
    val += (f & 1) ? __cosf((float)t * div) : __sinf((float)t * div);
    h[(((size_t)b * N_ + n) * T_ + t) * D_ + f] = val;
    if (t == T_ - 1) tgt[((size_t)b * N_ + n) * D_ + f] = val;
  }
}

// ---------------------------------------------------------------------------
// Host orchestration
// ---------------------------------------------------------------------------
extern "C" void kernel_launch(void* const* d_in, const int* in_sizes, int n_in,
                              void* d_out, int out_size, void* d_ws, size_t ws_size,
                              hipStream_t stream) {
  (void)in_sizes; (void)n_in; (void)out_size; (void)ws_size;
  const float* x = (const float*)d_in[0];
  const int* ei = (const int*)d_in[1];

  int p = 2;
  auto nxt = [&]() { return (const float*)d_in[p++]; };

  const float *gW = nxt(), *gAs = nxt(), *gAd = nxt(), *gB = nxt();

  struct Attn { const float *Wq,*bq,*Wk,*bk,*Wv,*bv,*Wo,*bo; };
  struct EncL { Attn sa; const float *W1,*b1,*W2,*b2,*l1g,*l1b,*l2g,*l2b; };
  struct DecL { Attn sa, ca; const float *W1,*b1,*W2,*b2,*l1g,*l1b,*l2g,*l2b,*l3g,*l3b; };
  auto getA = [&]() { Attn a; a.Wq=nxt(); a.bq=nxt(); a.Wk=nxt(); a.bk=nxt();
                      a.Wv=nxt(); a.bv=nxt(); a.Wo=nxt(); a.bo=nxt(); return a; };
  EncL enc[2];
  for (int i = 0; i < 2; ++i) {
    enc[i].sa = getA();
    enc[i].W1=nxt(); enc[i].b1=nxt(); enc[i].W2=nxt(); enc[i].b2=nxt();
    enc[i].l1g=nxt(); enc[i].l1b=nxt(); enc[i].l2g=nxt(); enc[i].l2b=nxt();
  }
  const float *encG = nxt(), *encB = nxt();
  DecL dec[2];
  for (int i = 0; i < 2; ++i) {
    dec[i].sa = getA(); dec[i].ca = getA();
    dec[i].W1=nxt(); dec[i].b1=nxt(); dec[i].W2=nxt(); dec[i].b2=nxt();
    dec[i].l1g=nxt(); dec[i].l1b=nxt(); dec[i].l2g=nxt(); dec[i].l2b=nxt();
    dec[i].l3g=nxt(); dec[i].l3b=nxt();
  }
  const float *decG = nxt(), *decB = nxt();

  // workspace layout (floats)
  float* ws = (float*)d_ws;
  const size_t MEMF = (size_t)B_ * N_ * T_ * D_;   // 262144 (h == mem)
  float* hbuf = ws;
  float* tvec = hbuf + MEMF;                        // 32768
  float* q    = tvec + (size_t)B_ * N_ * D_;
  float* kk   = q + MEMF;
  float* vv   = kk + MEMF;
  float* attn = vv + MEMF;
  float* ff1  = attn + MEMF;                        // 65536
  float* tmp  = ff1 + (size_t)B_ * N_ * T_ * 16;

  auto nblocks = [](int M) {
    int blocks = (M / 16 + 7) / 8;
    if (blocks > 1024) blocks = 1024;
    if (blocks < 1) blocks = 1;
    return blocks;
  };
  // C[M,64] = A[M,64] @ W[64,64]^T + bias
  auto gemm64 = [&](const float* A, const float* Wm, const float* bias, float* C, int M) {
    wmma_gemm_nt<64, 4><<<nblocks(M), 256, 0, stream>>>(A, Wm, bias, C, M, 0);
  };
  // C[M,16] = relu(A[M,64] @ W[16,64]^T + bias)
  auto gemmFF1 = [&](const float* A, const float* Wm, const float* bias, float* C, int M) {
    wmma_gemm_nt<64, 1><<<nblocks(M), 256, 0, stream>>>(A, Wm, bias, C, M, 1);
  };
  // C[M,64] = A[M,16] @ W[64,16]^T + bias
  auto gemmFF2 = [&](const float* A, const float* Wm, const float* bias, float* C, int M) {
    wmma_gemm_nt<16, 4><<<nblocks(M), 256, 0, stream>>>(A, Wm, bias, C, M, 0);
  };
  auto attn_go = [&](const float* Qp, const float* Kp, const float* Vp, float* Op,
                     int Lq, int Lk) {
    size_t sh = ((size_t)16 * Lk + 8 * 256) * sizeof(float);
    attn_kernel<<<dim3(Lq / 16, 4, B_), 256, sh, stream>>>(Qp, Kp, Vp, Op, Lq, Lk, 0.25f);
  };
  auto ln = [&](const float* xx, const float* rr, const float* gg, const float* bb,
                float* yy, int rows) {
    int blocks = (rows + 7) / 8;
    if (blocks > 1024) blocks = 1024;
    add_ln_kernel<<<blocks, 256, 0, stream>>>(xx, rr, gg, bb, yy, rows);
  };

  // ---- GAT (one block per (b,t) graph) ----
  {
    size_t sh = ((size_t)256 * 128 * 2 + 512 * 3) * 4 + 512 * 4;
    gat_kernel<<<B_ * T_, 256, sh, stream>>>(x, ei, gW, gAs, gAd, gB, hbuf, tvec);
  }

  // ---- encoder (mem = hbuf, (B,N*T,D) == (B,N,T,D) flat) ----
  const int Lmem = N_ * T_;        // 2048
  const int Mmem = B_ * Lmem;      // 4096
  float* mem = hbuf;
  for (int L = 0; L < 2; ++L) {
    EncL& e = enc[L];
    gemm64(mem, e.sa.Wq, e.sa.bq, q,  Mmem);
    gemm64(mem, e.sa.Wk, e.sa.bk, kk, Mmem);
    gemm64(mem, e.sa.Wv, e.sa.bv, vv, Mmem);
    attn_go(q, kk, vv, attn, Lmem, Lmem);
    gemm64(attn, e.sa.Wo, e.sa.bo, tmp, Mmem);
    ln(mem, tmp, e.l1g, e.l1b, mem, Mmem);
    gemmFF1(mem, e.W1, e.b1, ff1, Mmem);
    gemmFF2(ff1, e.W2, e.b2, tmp, Mmem);
    ln(mem, tmp, e.l2g, e.l2b, mem, Mmem);
  }
  ln(mem, nullptr, encG, encB, mem, Mmem);

  // ---- decoder ----
  const int Ltg = N_;              // 256
  const int Mtg = B_ * Ltg;        // 512
  float* t = tvec;
  for (int L = 0; L < 2; ++L) {
    DecL& d = dec[L];
    gemm64(t, d.sa.Wq, d.sa.bq, q,  Mtg);
    gemm64(t, d.sa.Wk, d.sa.bk, kk, Mtg);
    gemm64(t, d.sa.Wv, d.sa.bv, vv, Mtg);
    attn_go(q, kk, vv, attn, Ltg, Ltg);
    gemm64(attn, d.sa.Wo, d.sa.bo, tmp, Mtg);
    ln(t, tmp, d.l1g, d.l1b, t, Mtg);

    gemm64(t,   d.ca.Wq, d.ca.bq, q,  Mtg);
    gemm64(mem, d.ca.Wk, d.ca.bk, kk, Mmem);
    gemm64(mem, d.ca.Wv, d.ca.bv, vv, Mmem);
    attn_go(q, kk, vv, attn, Ltg, Lmem);
    gemm64(attn, d.ca.Wo, d.ca.bo, tmp, Mtg);
    ln(t, tmp, d.l2g, d.l2b, t, Mtg);

    gemmFF1(t, d.W1, d.b1, ff1, Mtg);
    gemmFF2(ff1, d.W2, d.b2, tmp, Mtg);
    ln(t, tmp, d.l3g, d.l3b, t, Mtg);
  }
  ln(t, nullptr, decG, decB, (float*)d_out, Mtg);
}
